// MultiHeadedLocalAttention_82841329206089
// MI455X (gfx1250) — matrix operand order repaired and checked by reading, not compiled, not run on hardware
//
#include <hip/hip_runtime.h>
#include <hip/hip_bf16.h>

// ---------------- problem constants ----------------
#define B_    16
#define HW_   56
#define DIM_  384
#define NH_   12
#define DH_   32
#define WIN_  7
#define WT_   49          // tokens per window (7*7)
#define WPAD_ 64          // padded tokens per window
#define WS_   8           // windows per side
#define NWIN_ 64          // windows per image
#define XSTRIDE (DIM_ + 8)   // LDS row stride (f16) for 384-wide tiles

typedef _Float16 v16h __attribute__((ext_vector_type(16)));
typedef _Float16 v8h  __attribute__((ext_vector_type(8)));
typedef float    v8f  __attribute__((ext_vector_type(8)));

static __device__ __forceinline__ v16h cat8(v8h a, v8h b) {
  return __builtin_shufflevector(a, b, 0,1,2,3,4,5,6,7,8,9,10,11,12,13,14,15);
}

// A fragment (16x32 f16, row-major source [M][K]):
// lanes 0-15: M=lane, K {k0..k0+7, k0+16..k0+23}; lanes 16-31: M=lane-16, K {k0+8..15, k0+24..31}
static __device__ __forceinline__ v16h load_a_frag(const _Float16* row0, int stride,
                                                   int lhalf, int hi) {
  const _Float16* p = row0 + lhalf * stride + (hi ? 8 : 0);
  return cat8(*(const v8h*)p, *(const v8h*)(p + 16));
}

// B fragment (32x16 f16), source stored transposed row-major [N][K]:
// lanes 0-15: N=lane, K {k0..k0+7, k0+16..k0+23}; lanes 16-31: N=lane-16, K {k0+8..15, k0+24..31}
static __device__ __forceinline__ v16h load_b_frag(const _Float16* col0, int stride,
                                                   int lhalf, int hi) {
  const _Float16* p = col0 + lhalf * stride + (hi ? 8 : 0);
  return cat8(*(const v8h*)p, *(const v8h*)(p + 16));
}

static __device__ __forceinline__ v8f wmma16(v16h a, v16h b, v8f c) {
  return __builtin_amdgcn_wmma_f32_16x16x32_f16(false, a, false, b, (short)0, c, false, false);
}

// ---------------- K0: weight fp32 -> f16 ----------------
__global__ __launch_bounds__(256) void convert_weights(const float* __restrict__ wqkv,
                                                       const float* __restrict__ wout,
                                                       _Float16* __restrict__ wqkv_h,
                                                       _Float16* __restrict__ wout_h) {
  int i = blockIdx.x * 256 + threadIdx.x;
  if (i < 3 * DIM_ * DIM_) wqkv_h[i] = (_Float16)wqkv[i];
  if (i < DIM_ * DIM_)     wout_h[i] = (_Float16)wout[i];
}

// ---------------- K1: roll + window gather + QKV GEMM ----------------
// qkvbuf layout: [part(3)][b][head][win][row(64)][d(32)], f16; rows 49..63 are zeros.
__global__ __launch_bounds__(256) void qkv_kernel(const float* __restrict__ x,
                                                  const _Float16* __restrict__ wq,
                                                  _Float16* __restrict__ qkvbuf) {
  __shared__ __align__(16) _Float16 xs[64 * XSTRIDE];

  const int blk = blockIdx.x;
  const int b   = blk >> 6;
  const int win = blk & 63;
  const int wi  = win >> 3, wj = win & 7;
  const int tid = threadIdx.x;

  // zero the A tile (pad rows -> exact zero outputs)
  {
    v8h hz = {0,0,0,0,0,0,0,0};
    for (int i = tid; i < 64 * XSTRIDE / 8; i += 256) ((v8h*)xs)[i] = hz;
  }
  __syncthreads();

  // gather 49 rolled rows, fp32 -> f16
  for (int i = tid; i < WT_ * (DIM_ / 4); i += 256) {
    int t = i / (DIM_ / 4), c = i % (DIM_ / 4);
    int ti = t / WIN_, tj = t % WIN_;
    int gi = (wi * WIN_ + ti + 3) % HW_;       // roll(-3): xs[si] = x[si+3]
    int gj = (wj * WIN_ + tj + 3) % HW_;
    const float4 f = *(const float4*)(x + (((size_t)b * HW_ + gi) * HW_ + gj) * DIM_ + c * 4);
    _Float16* dst = xs + t * XSTRIDE + c * 4;
    dst[0] = (_Float16)f.x; dst[1] = (_Float16)f.y;
    dst[2] = (_Float16)f.z; dst[3] = (_Float16)f.w;
  }
  __syncthreads();

  const int wave = tid >> 5, lane = tid & 31;
  const int lhalf = lane & 15, hi = lane >> 4;
  const v8f zf = {0,0,0,0,0,0,0,0};

  for (int nt = 0; nt < 9; ++nt) {             // 8 waves * 9 tiles * 16 = 1152 cols
    const int n0 = wave * 144 + nt * 16;
    v8f acc[4] = {zf, zf, zf, zf};
    for (int k0 = 0; k0 < DIM_; k0 += 32) {
      __builtin_prefetch(wq + (size_t)n0 * DIM_ + k0 + 64, 0, 1);
      v16h bf = load_b_frag(wq + (size_t)n0 * DIM_ + k0, DIM_, lhalf, hi);
#pragma unroll
      for (int mt = 0; mt < 4; ++mt) {
        v16h af = load_a_frag(xs + mt * 16 * XSTRIDE + k0, XSTRIDE, lhalf, hi);
        acc[mt] = wmma16(af, bf, acc[mt]);
      }
    }
    // epilogue: scatter into windowed layout, fold 1/sqrt(32) into Q
    const int n    = n0 + lhalf;
    const int part = n / DIM_;
    const int rem  = n - part * DIM_;
    const int head = rem >> 5, d = rem & 31;
    const float sc = (part == 0) ? 0.17677669529663689f : 1.0f;
    _Float16* dst = qkvbuf +
        ((((size_t)part * B_ + b) * NH_ + head) * NWIN_ + win) * (WPAD_ * DH_) + d;
#pragma unroll
    for (int mt = 0; mt < 4; ++mt)
#pragma unroll
      for (int v = 0; v < 8; ++v) {
        int m = mt * 16 + v + (hi ? 8 : 0);
        dst[m * DH_] = (_Float16)(acc[mt][v] * sc);
      }
  }
}

// ---------------- K2: window attention (one wave per b,head,window) ----------------
__global__ __launch_bounds__(32) void attn_kernel(const _Float16* __restrict__ qkvbuf,
                                                  const float* __restrict__ relt,
                                                  _Float16* __restrict__ attno) {
  __shared__ __align__(16) float    srel[169];
  __shared__ __align__(16) float    smem[64 * 68];
  __shared__ __align__(16) _Float16 pmem[64 * 72];
  __shared__ __align__(16) _Float16 vt[32 * 72];

  const int blk  = blockIdx.x;
  const int win  = blk & 63;
  const int th   = blk >> 6;
  const int head = th % NH_;
  const int b    = th / NH_;
  const int lane = threadIdx.x;
  const int lhalf = lane & 15, hi = lane >> 4;

  const size_t slab = (size_t)WPAD_ * DH_;
  const _Float16* qbase = qkvbuf + ((((size_t)0 * B_ + b) * NH_ + head) * NWIN_ + win) * slab;
  const _Float16* kbase = qkvbuf + ((((size_t)1 * B_ + b) * NH_ + head) * NWIN_ + win) * slab;
  const _Float16* vbase = qkvbuf + ((((size_t)2 * B_ + b) * NH_ + head) * NWIN_ + win) * slab;

  // stage relative-position bias column for this head
  for (int i = lane; i < 169; i += 32) srel[i] = relt[i * NH_ + head];

  // stage V transposed: vt[d][key]
  for (int r = lane; r < WPAD_; r += 32) {
    const v8h* src = (const v8h*)(vbase + r * DH_);
    v8h c0 = src[0], c1 = src[1], c2 = src[2], c3 = src[3];
#pragma unroll
    for (int d = 0; d < 8; ++d) {
      vt[(d     ) * 72 + r] = c0[d];
      vt[(d +  8) * 72 + r] = c1[d];
      vt[(d + 16) * 72 + r] = c2[d];
      vt[(d + 24) * 72 + r] = c3[d];
    }
  }

  // S = Q * K^T  (16 WMMAs, K-dim = 32 in one shot)
  v16h qa[4];
#pragma unroll
  for (int mt = 0; mt < 4; ++mt)
    qa[mt] = load_a_frag(qbase + mt * 16 * DH_, DH_, lhalf, hi);

  const v8f zf = {0,0,0,0,0,0,0,0};
  v8f s[4][4];
#pragma unroll
  for (int nt = 0; nt < 4; ++nt) {
    v16h kb = load_b_frag(kbase + nt * 16 * DH_, DH_, lhalf, hi);
#pragma unroll
    for (int mt = 0; mt < 4; ++mt)
      s[mt][nt] = wmma16(qa[mt], kb, zf);
  }

  // spill scores to LDS
#pragma unroll
  for (int mt = 0; mt < 4; ++mt)
#pragma unroll
    for (int nt = 0; nt < 4; ++nt)
#pragma unroll
      for (int v = 0; v < 8; ++v)
        smem[(mt * 16 + v + (hi ? 8 : 0)) * 68 + nt * 16 + lhalf] = s[mt][nt][v];
  __syncthreads();

  // softmax: rel bias + shift mask; LR mask is all-False in the reference, only UL applies
  const int needUL = ((win >> 3) == (WS_ - 1));
  for (int rr = 0; rr < 2; ++rr) {
    const int m = lane + rr * 32;
    float* row = smem + m * 68;
    const int  mi = m / WIN_, mj = m % WIN_;
    const bool mv = (m < WT_);
    float mx = -1e30f;
    for (int n = 0; n < 64; ++n) {
      float v = row[n];
      if (n < WT_ && mv) {
        int di = mi - n / WIN_ + (WIN_ - 1);
        int dj = mj - n % WIN_ + (WIN_ - 1);
        v += srel[di * (2 * WIN_ - 1) + dj];
        if (needUL && ((m >= 28) != (n >= 28))) v = -1e30f;
      }
      if (n >= WT_) v = -1e30f;
      row[n] = v;
      mx = fmaxf(mx, v);
    }
    float sum = 0.f;
    for (int n = 0; n < 64; ++n) { float e = __expf(row[n] - mx); row[n] = e; sum += e; }
    const float inv = 1.f / sum;
    _Float16* prow = pmem + m * 72;
    for (int n = 0; n < 64; ++n) prow[n] = (_Float16)(row[n] * inv);
  }
  __syncthreads();

  // O = P * V  (4x2 tiles, K = 64 -> two k-steps)
  v8f o[4][2];
#pragma unroll
  for (int mt = 0; mt < 4; ++mt) { o[mt][0] = zf; o[mt][1] = zf; }
#pragma unroll
  for (int ks = 0; ks < 2; ++ks) {
    const int k0 = ks * 32;
    v16h pa[4];
#pragma unroll
    for (int mt = 0; mt < 4; ++mt)
      pa[mt] = load_a_frag(pmem + mt * 16 * 72 + k0, 72, lhalf, hi);
#pragma unroll
    for (int nt = 0; nt < 2; ++nt) {
      v16h vb = load_b_frag(vt + nt * 16 * 72 + k0, 72, lhalf, hi);
#pragma unroll
      for (int mt = 0; mt < 4; ++mt)
        o[mt][nt] = wmma16(pa[mt], vb, o[mt][nt]);
    }
  }

  _Float16* obase = attno + (((size_t)b * NH_ + head) * NWIN_ + win) * slab;
#pragma unroll
  for (int mt = 0; mt < 4; ++mt)
#pragma unroll
    for (int nt = 0; nt < 2; ++nt)
#pragma unroll
      for (int v = 0; v < 8; ++v) {
        int m = mt * 16 + v + (hi ? 8 : 0);
        obase[m * DH_ + nt * 16 + lhalf] = (_Float16)o[mt][nt][v];
      }
}

// ---------------- K3: head merge + output GEMM + bias + inverse roll ----------------
__global__ __launch_bounds__(256) void proj_kernel(const _Float16* __restrict__ attno,
                                                   const _Float16* __restrict__ wout_h,
                                                   const float* __restrict__ bias,
                                                   float* __restrict__ out) {
  __shared__ __align__(16) _Float16 as[64 * XSTRIDE];

  const int blk = blockIdx.x;
  const int b   = blk >> 6;
  const int win = blk & 63;
  const int wi  = win >> 3, wj = win & 7;
  const int tid = threadIdx.x;

  {
    v8h hz = {0,0,0,0,0,0,0,0};
    for (int i = tid; i < 64 * XSTRIDE / 8; i += 256) ((v8h*)as)[i] = hz;
  }
  __syncthreads();

  for (int i = tid; i < WT_ * NH_; i += 256) {
    int t = i / NH_, h = i % NH_;
    const v8h* src = (const v8h*)(attno +
        ((((size_t)b * NH_ + h) * NWIN_ + win) * WPAD_ + t) * DH_);
    v8h* dst = (v8h*)(as + t * XSTRIDE + h * DH_);
    dst[0] = src[0]; dst[1] = src[1]; dst[2] = src[2]; dst[3] = src[3];
  }
  __syncthreads();

  const int wave = tid >> 5, lane = tid & 31;
  const int lhalf = lane & 15, hi = lane >> 4;
  const v8f zf = {0,0,0,0,0,0,0,0};

  for (int nt = 0; nt < 3; ++nt) {             // 8 waves * 3 tiles * 16 = 384 cols
    const int n0 = wave * 48 + nt * 16;
    v8f acc[4] = {zf, zf, zf, zf};
    for (int k0 = 0; k0 < DIM_; k0 += 32) {
      v16h bf = load_b_frag(wout_h + (size_t)n0 * DIM_ + k0, DIM_, lhalf, hi);
#pragma unroll
      for (int mt = 0; mt < 4; ++mt) {
        v16h af = load_a_frag(as + mt * 16 * XSTRIDE + k0, XSTRIDE, lhalf, hi);
        acc[mt] = wmma16(af, bf, acc[mt]);
      }
    }
    const int n = n0 + lhalf;
    const float bn = bias[n];
#pragma unroll
    for (int mt = 0; mt < 4; ++mt)
#pragma unroll
      for (int v = 0; v < 8; ++v) {
        int m = mt * 16 + v + (hi ? 8 : 0);
        if (m < WT_) {
          int ti = m / WIN_, tj = m % WIN_;
          int gi = (wi * WIN_ + ti + 3) % HW_;   // roll(+3): out[si+3] = proj[si]
          int gj = (wj * WIN_ + tj + 3) % HW_;
          out[(((size_t)b * HW_ + gi) * HW_ + gj) * DIM_ + n] = acc[mt][v] + bn;
        }
      }
  }
}

// ---------------- launch ----------------
extern "C" void kernel_launch(void* const* d_in, const int* in_sizes, int n_in,
                              void* d_out, int out_size, void* d_ws, size_t ws_size,
                              hipStream_t stream) {
  (void)in_sizes; (void)n_in; (void)out_size; (void)ws_size;
  const float* x    = (const float*)d_in[0];
  const float* wqkv = (const float*)d_in[1];
  const float* wout = (const float*)d_in[2];
  const float* bout = (const float*)d_in[3];
  const float* relt = (const float*)d_in[4];
  float* out = (float*)d_out;

  char* ws = (char*)d_ws;
  const size_t off_wqkv = 0;
  const size_t off_wout = off_wqkv + (size_t)3 * DIM_ * DIM_ * 2;     //   884,736
  const size_t off_qkv  = off_wout + (size_t)DIM_ * DIM_ * 2;         // 1,179,648
  const size_t off_attn = off_qkv +
      (size_t)3 * B_ * NH_ * NWIN_ * WPAD_ * DH_ * 2;                 // +150,994,944
  _Float16* wqkv_h = (_Float16*)(ws + off_wqkv);
  _Float16* wout_h = (_Float16*)(ws + off_wout);
  _Float16* qkvbuf = (_Float16*)(ws + off_qkv);
  _Float16* attno  = (_Float16*)(ws + off_attn);

  convert_weights<<<(3 * DIM_ * DIM_ + 255) / 256, 256, 0, stream>>>(wqkv, wout, wqkv_h, wout_h);
  qkv_kernel<<<B_ * NWIN_, 256, 0, stream>>>(x, wqkv_h, qkvbuf);
  attn_kernel<<<B_ * NH_ * NWIN_, 32, 0, stream>>>(qkvbuf, relt, attno);
  proj_kernel<<<B_ * NWIN_, 256, 0, stream>>>(attno, wout_h, bout, out);
}
